// MaskedHeteroGAT_84000970375723
// MI455X (gfx1250) — matrix operand order repaired and checked by reading, not compiled, not run on hardware
//
#include <hip/hip_runtime.h>
#include <hip/hip_bf16.h>

// ---- problem constants (from reference) ----
#define H_    4
#define C_    64
#define F_    256      // H_*C_
#define T_    6
#define DE_   8
#define KCL   48       // T_*DE_
#define DIN   128

typedef __attribute__((ext_vector_type(16))) __bf16 v16bf;
typedef __attribute__((ext_vector_type(8)))  __bf16 v8bf;
typedef __attribute__((ext_vector_type(4)))  __bf16 v4bf;
typedef __attribute__((ext_vector_type(8)))  float  v8f;

// -------- monotonic float<->uint encoding for atomicMax on floats --------
__device__ __forceinline__ unsigned int encF(float f) {
    unsigned int b = __float_as_uint(f);
    return (b & 0x80000000u) ? ~b : (b | 0x80000000u);
}
__device__ __forceinline__ float decF(unsigned int u) {
    unsigned int b = (u & 0x80000000u) ? (u ^ 0x80000000u) : ~u;
    return __uint_as_float(b);
}
#define ENC_NEG_INF 0x007FFFFFu   // encF(-inf)

// ======================= small utility kernels =======================
__global__ void fill_f32_kernel(float* p, float v, long n) {
    long i = (long)blockIdx.x * blockDim.x + threadIdx.x;
    if (i < n) p[i] = v;
}
__global__ void fill_u32_kernel(unsigned int* p, unsigned int v, long n) {
    long i = (long)blockIdx.x * blockDim.x + threadIdx.x;
    if (i < n) p[i] = v;
}
__global__ void init_bias_kernel(float* out, const float* __restrict__ b, long n) {
    long i = (long)blockIdx.x * blockDim.x + threadIdx.x;
    if (i < n) out[i] = b[i & (F_ - 1)];
}
__global__ void relu_kernel(float* p, long n) {
    long i = (long)blockIdx.x * blockDim.x + threadIdx.x;
    if (i < n) p[i] = fmaxf(p[i], 0.0f);
}

// ======================= bf16 WMMA GEMM =======================
// C[M,Ncols] = (rowScale ? diag(rowScale) : I) * A[M,K] @ B[K,Ncols]
// Block tile 64x128, 8 waves (4 along M x 2 along N), wave tile 16x64.
// Tiles staged in LDS as bf16, pre-swizzled to WMMA lane layout so each
// A fragment is a contiguous 32B LDS read, B fragment two 16B reads (2x ds_load_b128, zero inner cvts).
#define BM 64
#define BN 128
#define BK 32
#define BSTR 40   // ldsBt per-column k-stride (bf16 elems); pad breaks bank period

template<int KT>
__global__ __launch_bounds__(256)
void wmma_gemm_bf16(const float* __restrict__ A, const float* __restrict__ B,
                    float* __restrict__ C, const float* __restrict__ rowScale,
                    int M, int Ncols)
{
    // A frags pre-swizzled: [mt(4)][lane(32)][e(16)] bf16  = 4 KB
    __shared__ __align__(16) __bf16 ldsA[4 * 32 * 16];
    // B transposed: [col(128)][k(32) padded to 40] bf16    = 10 KB
    __shared__ __align__(16) __bf16 ldsBt[BN * BSTR];

    const int tid    = threadIdx.x;
    const int lane   = tid & 31;
    const int wave   = tid >> 5;
    const int waveM  = wave & 3;
    const int waveN  = wave >> 2;
    const int blockM = blockIdx.x * BM;
    const int blockN = blockIdx.y * BN;

    // ---- staging coordinates (constant across K steps) ----
    // A: one thread converts 8 consecutive k of one row -> one b128 store.
    const int ar   = tid >> 2;                 // tile row 0..63
    const int ag8  = tid & 3;                  // k-octet 0..3
    const int aGr  = blockM + ar;
    const float aScale = (rowScale && aGr < M) ? rowScale[aGr] : 1.0f;
    // inverse of ISA layout: lane = (r&15) + 16*((k>>3)&1), e = 8*(k>>4) + (k&7)
    __bf16* aDst = ldsA + (((ar >> 4) * 32 + (ar & 15) + 16 * (ag8 & 1)) * 16
                           + 8 * (ag8 >> 1));
    // B: one thread transposes a 4k x 4col block -> 4 b64 column stores.
    const int bcg = (tid & 31) * 4;            // first col of group
    const int bkg = (tid >> 5) * 4;            // first k of group

    v8f acc[4];
    {
        v8f zero = {0.f,0.f,0.f,0.f,0.f,0.f,0.f,0.f};
        #pragma unroll
        for (int i = 0; i < 4; ++i) acc[i] = zero;
    }

    #pragma unroll
    for (int k0 = 0; k0 < KT; k0 += BK) {
        // ---- stage A (convert + swizzle) ----
        {
            v8bf av;
            #pragma unroll
            for (int e = 0; e < 8; ++e) av[e] = (__bf16)0.0f;
            if (aGr < M) {
                const float* ap = A + (size_t)aGr * KT + k0 + ag8 * 8;
                float4 v0 = *(const float4*)ap;
                float4 v1 = *(const float4*)(ap + 4);
                av[0] = (__bf16)(v0.x * aScale); av[1] = (__bf16)(v0.y * aScale);
                av[2] = (__bf16)(v0.z * aScale); av[3] = (__bf16)(v0.w * aScale);
                av[4] = (__bf16)(v1.x * aScale); av[5] = (__bf16)(v1.y * aScale);
                av[6] = (__bf16)(v1.z * aScale); av[7] = (__bf16)(v1.w * aScale);
                if (k0 + BK < KT) __builtin_prefetch(ap + BK, 0, 3);
            }
            *(v8bf*)aDst = av;
        }
        // ---- stage B (convert + transpose) ----
        {
            const float* bp = B + (size_t)(k0 + bkg) * Ncols + blockN + bcg;
            float4 r0 = *(const float4*)bp;
            float4 r1 = *(const float4*)(bp + Ncols);
            float4 r2 = *(const float4*)(bp + 2 * (size_t)Ncols);
            float4 r3 = *(const float4*)(bp + 3 * (size_t)Ncols);
            if (k0 + BK < KT) __builtin_prefetch(bp + (size_t)BK * Ncols, 0, 3);
            v4bf c0, c1, c2, c3;
            c0[0]=(__bf16)r0.x; c0[1]=(__bf16)r1.x; c0[2]=(__bf16)r2.x; c0[3]=(__bf16)r3.x;
            c1[0]=(__bf16)r0.y; c1[1]=(__bf16)r1.y; c1[2]=(__bf16)r2.y; c1[3]=(__bf16)r3.y;
            c2[0]=(__bf16)r0.z; c2[1]=(__bf16)r1.z; c2[2]=(__bf16)r2.z; c2[3]=(__bf16)r3.z;
            c3[0]=(__bf16)r0.w; c3[1]=(__bf16)r1.w; c3[2]=(__bf16)r2.w; c3[3]=(__bf16)r3.w;
            *(v4bf*)(ldsBt + (bcg + 0) * BSTR + bkg) = c0;
            *(v4bf*)(ldsBt + (bcg + 1) * BSTR + bkg) = c1;
            *(v4bf*)(ldsBt + (bcg + 2) * BSTR + bkg) = c2;
            *(v4bf*)(ldsBt + (bcg + 3) * BSTR + bkg) = c3;
        }
        __syncthreads();

        // ---- A fragment: contiguous 32B ----
        v16bf afrag;
        {
            const __bf16* aPtr = ldsA + (waveM * 32 + lane) * 16;
            v8bf a0 = *(const v8bf*)aPtr;
            v8bf a1 = *(const v8bf*)(aPtr + 8);
            #pragma unroll
            for (int e = 0; e < 8; ++e) { afrag[e] = a0[e]; afrag[8 + e] = a1[e]; }
        }
        // ---- 4 B fragments + WMMA ----
        #pragma unroll
        for (int nt = 0; nt < 4; ++nt) {
            const int col = waveN * 64 + nt * 16 + (lane & 15);
            const __bf16* bPtr = ldsBt + col * BSTR + (lane >> 4) * 8;
            v8bf b0 = *(const v8bf*)bPtr;
            v8bf b1 = *(const v8bf*)(bPtr + 16);
            v16bf bfrag;
            #pragma unroll
            for (int e = 0; e < 8; ++e) { bfrag[e] = b0[e]; bfrag[8 + e] = b1[e]; }
            acc[nt] = __builtin_amdgcn_wmma_f32_16x16x32_bf16(
                false, afrag, false, bfrag, (short)0, acc[nt], false, false);
        }
        __syncthreads();
    }

    // ---- store C : VGPR i -> row i (+8 for lanes 16..31), col = lane&15 ----
    const int rbase = blockM + waveM * 16 + ((lane >> 4) ? 8 : 0);
    const int cbase = blockN + waveN * 64 + (lane & 15);
    #pragma unroll
    for (int nt = 0; nt < 4; ++nt) {
        #pragma unroll
        for (int i = 0; i < 8; ++i) {
            int r = rbase + i;
            if (r < M) C[(size_t)r * Ncols + cbase + nt * 16] = acc[nt][i];
        }
    }
}

// ======================= GATv2 edge phases =======================
__global__ __launch_bounds__(256)
void edge_logits_kernel(const float* __restrict__ hs, const float* __restrict__ hd,
                        const int* __restrict__ src, const int* __restrict__ dst,
                        const float* __restrict__ att, float* __restrict__ logits,
                        unsigned int* __restrict__ mEnc, int E)
{
    __shared__ float red[256];
    const int tid = threadIdx.x;
    for (int e = blockIdx.x; e < E; e += gridDim.x) {
        int s = src[e], d = dst[e];
        float m = hs[(size_t)s * F_ + tid] + hd[(size_t)d * F_ + tid];
        m = (m > 0.f) ? m : 0.2f * m;
        red[tid] = m * att[tid];
        __syncthreads();
        #pragma unroll
        for (int off = 32; off > 0; off >>= 1) {
            if ((tid & 63) < off) red[tid] += red[tid + off];
            __syncthreads();
        }
        if ((tid & 63) == 0) {
            int h = tid >> 6;
            float lg = red[tid];
            logits[(size_t)e * H_ + h] = lg;
            atomicMax(&mEnc[(size_t)d * H_ + h], encF(lg));
        }
        __syncthreads();
    }
}

__global__ void edge_expsum_kernel(float* __restrict__ w,
                                   const unsigned int* __restrict__ mEnc,
                                   float* __restrict__ sums,
                                   const int* __restrict__ dst, long n)
{
    long i = (long)blockIdx.x * blockDim.x + threadIdx.x;
    if (i >= n) return;
    int  h = (int)(i & (H_ - 1));
    long e = i >> 2;
    int  d = dst[e];
    float mv  = decF(mEnc[(size_t)d * H_ + h]);
    float val = __expf(w[i] - mv);
    w[i] = val;
    atomicAdd(&sums[(size_t)d * H_ + h], val);
}

__global__ __launch_bounds__(256)
void edge_agg_kernel(const float* __restrict__ w, const float* __restrict__ sums,
                     const float* __restrict__ hs, const int* __restrict__ src,
                     const int* __restrict__ dst, float* __restrict__ out, int E)
{
    const int tid = threadIdx.x;
    const int h = tid >> 6;
    for (int e = blockIdx.x; e < E; e += gridDim.x) {
        int s = src[e], d = dst[e];
        float a = w[(size_t)e * H_ + h] / (sums[(size_t)d * H_ + h] + 1e-16f);
        atomicAdd(&out[(size_t)d * F_ + tid], a * hs[(size_t)s * F_ + tid]);
    }
}

// ======================= DiffPool / classifier =======================
__global__ __launch_bounds__(64)
void pool_assign_kernel(const float* __restrict__ h, const float* __restrict__ Wp,
                        const float* __restrict__ bp, float* __restrict__ sA, int N)
{
    __shared__ float hrow[F_];
    __shared__ float sc[KCL];
    __shared__ float mx, sm;
    int n = blockIdx.x;
    if (n >= N) return;
    int tid = threadIdx.x;
    for (int i = tid; i < F_; i += 64) hrow[i] = h[(size_t)n * F_ + i];
    __syncthreads();
    if (tid < KCL) {
        float acc = bp[tid];
        for (int f = 0; f < F_; ++f) acc += hrow[f] * Wp[f * KCL + tid];
        sc[tid] = acc;
    }
    __syncthreads();
    if (tid == 0) {
        float m = -1e30f;
        for (int k = 0; k < KCL; ++k) m = fmaxf(m, sc[k]);
        float s = 0.f;
        for (int k = 0; k < KCL; ++k) s += __expf(sc[k] - m);
        mx = m; sm = s;
    }
    __syncthreads();
    if (tid < KCL) sA[(size_t)n * KCL + tid] = __expf(sc[tid] - mx) / sm;
}

__global__ __launch_bounds__(256)
void pool_accum_kernel(const float* __restrict__ sA, const float* __restrict__ h,
                       float* __restrict__ xP, int N)
{
    __shared__ float sr[KCL];
    const int tid = threadIdx.x;   // owns column f = tid
    float acc[KCL];
    #pragma unroll
    for (int k = 0; k < KCL; ++k) acc[k] = 0.f;
    for (int n = blockIdx.x; n < N; n += gridDim.x) {
        __syncthreads();
        if (tid < KCL) sr[tid] = sA[(size_t)n * KCL + tid];
        __syncthreads();
        float hv = h[(size_t)n * F_ + tid];
        #pragma unroll
        for (int k = 0; k < KCL; ++k) acc[k] += sr[k] * hv;
    }
    #pragma unroll
    for (int k = 0; k < KCL; ++k) atomicAdd(&xP[k * F_ + tid], acc[k]);
}

__global__ __launch_bounds__(256)
void agg_edge_kernel(const float* __restrict__ ea, float* __restrict__ agg, long E)
{
    __shared__ float red[256];
    int b = blockIdx.x;            // 0..47 == t*8+d
    int t = b >> 3, d = b & 7;
    const float* base = ea + (size_t)t * E * DE_ + d;
    float s = 0.f;
    for (long e = threadIdx.x; e < E; e += 256) s += base[e * DE_];
    red[threadIdx.x] = s;
    __syncthreads();
    #pragma unroll
    for (int off = 128; off > 0; off >>= 1) {
        if (threadIdx.x < off) red[threadIdx.x] += red[threadIdx.x + off];
        __syncthreads();
    }
    if (threadIdx.x == 0) agg[b] = red[0] / (float)E;
}

__global__ __launch_bounds__(64)
void final_cls_kernel(const float* __restrict__ xP, const float* __restrict__ agg,
                      const float* __restrict__ Wc, const float* __restrict__ bc,
                      float* __restrict__ out)
{
    int k = threadIdx.x;
    if (k >= KCL) return;
    float acc = bc[0];
    for (int f = 0; f < F_; ++f) acc += xP[k * F_ + f] * Wc[f];
    acc += agg[k] * Wc[F_];
    out[k] = 1.f / (1.f + __expf(-acc));
}

// ======================= launcher =======================
extern "C" void kernel_launch(void* const* d_in, const int* in_sizes, int n_in,
                              void* d_out, int out_size, void* d_ws, size_t ws_size,
                              hipStream_t stream)
{
    (void)n_in; (void)out_size; (void)ws_size;
    const int  N    = in_sizes[3];
    const int  Npkg = in_sizes[0] / DIN;
    const long E    = (long)in_sizes[16] / (T_ * 2);
    const int  t    = T_ - 1;   // only outs[-1] reaches the output

    const float* x_pkg   = (const float*)d_in[0];
    const float* x_dst5  = (const float*)d_in[1]  + (size_t)t * N * DIN;
    const float* e_attr  = (const float*)d_in[2];
    const float* mask    = (const float*)d_in[3];
    const float* W1s     = (const float*)d_in[4]  + (size_t)t * DIN * F_;
    const float* W1d     = (const float*)d_in[5]  + (size_t)t * DIN * F_;
    const float* att1    = (const float*)d_in[6]  + (size_t)t * H_ * C_;
    const float* b1      = (const float*)d_in[7]  + (size_t)t * F_;
    const float* W2s     = (const float*)d_in[8]  + (size_t)t * DIN * F_;
    const float* W2d     = (const float*)d_in[9]  + (size_t)t * F_ * F_;
    const float* att2    = (const float*)d_in[10] + (size_t)t * H_ * C_;
    const float* b2      = (const float*)d_in[11] + (size_t)t * F_;
    const float* W_pool  = (const float*)d_in[12];
    const float* b_pool  = (const float*)d_in[13];
    const float* W_cls   = (const float*)d_in[14];
    const float* b_cls   = (const float*)d_in[15];
    const int*   src     = (const int*)d_in[16] + (size_t)t * 2 * E;
    const int*   dst     = src + E;

    // ---- workspace carve (floats) ----
    float* ws = (float*)d_ws;
    size_t o = 0;
    float*        bufA   = ws + o; o += (size_t)N * F_;   // hs1 / hs2
    float*        bufB   = ws + o; o += (size_t)N * F_;   // hd1 / hd2
    float*        bufC   = ws + o; o += (size_t)N * F_;   // h1  / h2
    float*        wbuf   = ws + o; o += (size_t)E * H_;   // logits -> softmax weights
    unsigned int* mEnc   = (unsigned int*)(ws + o); o += (size_t)N * H_;
    float*        sumB   = ws + o; o += (size_t)N * H_;
    float*        sAsgn  = ws + o; o += (size_t)N * KCL;
    float*        xPool  = ws + o; o += (size_t)KCL * F_;
    float*        aggE   = ws + o; o += KCL;

    const long NH = (long)N * H_;
    const long NF = (long)N * F_;
    const int  EB = 4096;               // edge grid-stride blocks

    auto gemm = [&](const float* A, const float* Bm, float* Cp, const float* scale,
                    int M, int Kd) {
        dim3 grid((M + BM - 1) / BM, F_ / BN);
        if (Kd == DIN)
            wmma_gemm_bf16<DIN><<<grid, 256, 0, stream>>>(A, Bm, Cp, scale, M, F_);
        else
            wmma_gemm_bf16<F_><<<grid, 256, 0, stream>>>(A, Bm, Cp, scale, M, F_);
    };
    auto gat_layer = [&](const float* hs, const float* hd, const float* att,
                         const float* bias, float* out) {
        fill_u32_kernel<<<(NH + 255) / 256, 256, 0, stream>>>(mEnc, ENC_NEG_INF, NH);
        fill_f32_kernel<<<(NH + 255) / 256, 256, 0, stream>>>(sumB, 0.f, NH);
        edge_logits_kernel<<<EB, 256, 0, stream>>>(hs, hd, src, dst, att, wbuf, mEnc, (int)E);
        long EH = E * H_;
        edge_expsum_kernel<<<(EH + 255) / 256, 256, 0, stream>>>(wbuf, mEnc, sumB, dst, EH);
        init_bias_kernel<<<(NF + 255) / 256, 256, 0, stream>>>(out, bias, NF);
        edge_agg_kernel<<<EB, 256, 0, stream>>>(wbuf, sumB, hs, src, dst, out, (int)E);
    };

    // ---- layer 1: h1 = relu(GATv2(xp, xd5)) ----
    gemm(x_pkg,  W1s, bufA, mask, Npkg, DIN);   // hs1 = (xp*mask) @ W1_src
    gemm(x_dst5, W1d, bufB, mask, N,    DIN);   // hd1 = (xd*mask) @ W1_dst
    gat_layer(bufA, bufB, att1, b1, bufC);
    relu_kernel<<<(NF + 255) / 256, 256, 0, stream>>>(bufC, NF);

    // ---- layer 2: h2 = GATv2(xp, h1) ----
    gemm(x_pkg, W2s, bufA, mask,    Npkg, DIN); // hs2
    gemm(bufC,  W2d, bufB, nullptr, N,    F_);  // hd2 = h1 @ W2_dst
    gat_layer(bufA, bufB, att2, b2, bufC);      // h2 (bufC reused after hd2 built)

    // ---- DiffPool + classifier ----
    pool_assign_kernel<<<N, 64, 0, stream>>>(bufC, W_pool, b_pool, sAsgn, N);
    fill_f32_kernel<<<(KCL * F_ + 255) / 256, 256, 0, stream>>>(xPool, 0.f, KCL * F_);
    pool_accum_kernel<<<512, 256, 0, stream>>>(sAsgn, bufC, xPool, N);
    agg_edge_kernel<<<KCL, 256, 0, stream>>>(e_attr, aggE, E);
    final_cls_kernel<<<1, 64, 0, stream>>>(xPool, aggE, W_cls, b_cls, (float*)d_out);
}
